// Model_34935263986116
// MI455X (gfx1250) — matrix operand order repaired and checked by reading, not compiled, not run on hardware
//
#include <hip/hip_runtime.h>

#define N_NODES   50000
#define N_EDGES   800000
#define N_GRAPHS  256
#define CHANNELS  64
#define CONVS     8
#define GIN_H     256
#define HIDDEN    1024
#define XCAT_W    ((CONVS + 1) * CHANNELS)   // 576
#define M_TILES   (N_NODES / 16)             // 3125

typedef __attribute__((ext_vector_type(16))) __bf16 v16bf;
typedef __attribute__((ext_vector_type(8)))  float  v8f;

union BF16x16 {
    v16bf v;
    unsigned int u[8];
};

// round-to-nearest-even f32 -> bf16 (scalar)
static __device__ __forceinline__ unsigned short f2bf(float f) {
    unsigned int b = __float_as_uint(f);
    unsigned int r = b + 0x7FFFu + ((b >> 16) & 1u);
    return (unsigned short)(r >> 16);
}

// pack two f32 into one dword of bf16 pairs (RNE) using v_perm_b32:
// result = { bf16(y), bf16(x) }  (low half = x)
static __device__ __forceinline__ unsigned int pack_bf16x2(float x, float y) {
    unsigned int bx = __float_as_uint(x);
    unsigned int by = __float_as_uint(y);
    bx += 0x7FFFu + ((bx >> 16) & 1u);
    by += 0x7FFFu + ((by >> 16) & 1u);
    // dst bytes (LSB first): bx[2], bx[3], by[2], by[3]
    return __builtin_amdgcn_perm(by, bx, 0x07060302u);
}

// ---------------------------------------------------------------------------
// Pre-pack hidden_w / out_w (fp32) into bf16 tiles laid out exactly as the
// WMMA B operand: lane half h, VGPR v -> K=(v>>2)*16 + 8h + 2(v&3) (+0,+1).
// Index: ((conv*NT+nt)*KT+kt)*256 + lane*8 + v
// ---------------------------------------------------------------------------
__global__ void pack_weights_kernel(const float* __restrict__ hw,
                                    const float* __restrict__ ow,
                                    unsigned int* __restrict__ w1p,
                                    unsigned int* __restrict__ w2p) {
    int idx = blockIdx.x * blockDim.x + threadIdx.x;
    const int W1 = CONVS * 16 * 4 * 256;   // 131072
    const int W2 = CONVS * 4 * 8 * 256;    // 65536
    if (idx < W1) {
        int v    = idx & 7;
        int lane = (idx >> 3) & 31;
        int kt   = (idx >> 8) & 3;
        int nt   = (idx >> 10) & 15;
        int conv = idx >> 14;
        int n = nt * 16 + (lane & 15);
        int k = kt * 32 + (v >> 2) * 16 + (lane >> 4) * 8 + 2 * (v & 3);
        const float* w = hw + (size_t)conv * (2 * CHANNELS) * GIN_H;
        w1p[idx] = pack_bf16x2(w[(size_t)k * GIN_H + n], w[(size_t)(k + 1) * GIN_H + n]);
    } else if (idx - W1 < W2) {
        int j    = idx - W1;
        int v    = j & 7;
        int lane = (j >> 3) & 31;
        int kt   = (j >> 8) & 7;
        int nt   = (j >> 11) & 3;
        int conv = j >> 13;
        int n = nt * 16 + (lane & 15);
        int k = kt * 32 + (v >> 2) * 16 + (lane >> 4) * 8 + 2 * (v & 3);
        const float* w = ow + (size_t)conv * GIN_H * CHANNELS;
        w2p[j] = pack_bf16x2(w[(size_t)k * CHANNELS + n], w[(size_t)(k + 1) * CHANNELS + n]);
    }
}

// x0 = emb[node_types]  -> column block 0 of xcat
__global__ void embed_kernel(const int* __restrict__ node_types,
                             const float* __restrict__ emb,
                             float* __restrict__ xcat) {
    int idx = blockIdx.x * blockDim.x + threadIdx.x;
    if (idx >= N_NODES * CHANNELS) return;
    int n = idx >> 6, c = idx & 63;
    xcat[(size_t)n * XCAT_W + c] = emb[(size_t)node_types[n] * CHANNELS + c];
}

// Both scatter-sums: agg[:,0:64]  += x[src] at tgt ;  agg[:,64:128] += x[tgt] at src
__global__ void scatter_kernel(const int* __restrict__ ei,
                               const float* __restrict__ xcat,
                               float* __restrict__ agg, int conv) {
    int idx = blockIdx.x * blockDim.x + threadIdx.x;
    if (idx >= N_EDGES * 16) return;
    int e  = idx >> 4;
    int c0 = (idx & 15) * 4;
    int src = ei[e];
    int tgt = ei[N_EDGES + e];
    size_t xoff = (size_t)conv * CHANNELS + c0;
    float4 xs = *(const float4*)(xcat + (size_t)src * XCAT_W + xoff);
    float4 xt = *(const float4*)(xcat + (size_t)tgt * XCAT_W + xoff);
    float* o = agg + (size_t)tgt * 128 + c0;
    atomicAdd(o + 0, xs.x); atomicAdd(o + 1, xs.y);
    atomicAdd(o + 2, xs.z); atomicAdd(o + 3, xs.w);
    float* b = agg + (size_t)src * 128 + 64 + c0;
    atomicAdd(b + 0, xt.x); atomicAdd(b + 1, xt.y);
    atomicAdd(b + 2, xt.z); atomicAdd(b + 3, xt.w);
}

// ---------------------------------------------------------------------------
// Fused GIN MLP: per wave, 16 nodes:
//   h = relu([out||back] @ W1 + b1)   (128->256, bf16 WMMA, h kept in LDS)
//   c = h @ W2 + b2                   (256->64,  bf16 WMMA)
// Epilogue accumulates per-channel sum / sumsq for the instance norm.
// A-operand 16-bit layout (per lane: m = lane&15, half = lane>>4):
//   VGPR v in 0..3 -> K = kt*32 + half*8 + 2v      (contiguous 8 K values)
//   VGPR v in 4..7 -> K = kt*32 + 16 + half*8 + 2(v-4)
// so each (kt, quad) is one 16-byte contiguous load.
// ---------------------------------------------------------------------------
__global__ __launch_bounds__(128) void conv_gemm_kernel(
    const float* __restrict__ agg,
    const unsigned int* __restrict__ w1p,
    const unsigned int* __restrict__ w2p,
    const float* __restrict__ hidden_b,
    const float* __restrict__ out_b,
    float* __restrict__ cbuf,
    float* __restrict__ stats, int conv)
{
    __shared__ unsigned short hsh[4][16][GIN_H + 8];   // bf16 hidden, 528B row stride
    const int wave = threadIdx.x >> 5;
    const int lane = threadIdx.x & 31;
    const int tile = blockIdx.x * 4 + wave;
    if (tile >= M_TILES) return;                        // wave-uniform
    const int row0 = tile * 16;
    const int m    = lane & 15;
    const int half = lane >> 4;

    // ---- Load A (16x128 fp32 -> bf16) for all 4 K-steps of GEMM1 ----
    BF16x16 a[4];
    const float* arow = agg + (size_t)(row0 + m) * 128;
    #pragma unroll
    for (int kt = 0; kt < 4; ++kt) {
        const float4* p0 = (const float4*)(arow + kt * 32 + half * 8);       // K base..base+7
        const float4* p1 = (const float4*)(arow + kt * 32 + 16 + half * 8);  // K base+16..+23
        float4 f0 = p0[0], f1 = p0[1];
        float4 f2 = p1[0], f3 = p1[1];
        a[kt].u[0] = pack_bf16x2(f0.x, f0.y);
        a[kt].u[1] = pack_bf16x2(f0.z, f0.w);
        a[kt].u[2] = pack_bf16x2(f1.x, f1.y);
        a[kt].u[3] = pack_bf16x2(f1.z, f1.w);
        a[kt].u[4] = pack_bf16x2(f2.x, f2.y);
        a[kt].u[5] = pack_bf16x2(f2.z, f2.w);
        a[kt].u[6] = pack_bf16x2(f3.x, f3.y);
        a[kt].u[7] = pack_bf16x2(f3.z, f3.w);
    }

    // ---- GEMM1: 16 N-tiles x 4 K-steps -> relu(h) in LDS ----
    #pragma unroll
    for (int nt = 0; nt < 16; ++nt) {
        v8f acc = {0.f, 0.f, 0.f, 0.f, 0.f, 0.f, 0.f, 0.f};
        #pragma unroll
        for (int kt = 0; kt < 4; ++kt) {
            BF16x16 b;
            const uint4* p = (const uint4*)(w1p + ((((conv * 16 + nt) * 4 + kt) << 8) + lane * 8));
            uint4 q0 = p[0], q1 = p[1];
            b.u[0] = q0.x; b.u[1] = q0.y; b.u[2] = q0.z; b.u[3] = q0.w;
            b.u[4] = q1.x; b.u[5] = q1.y; b.u[6] = q1.z; b.u[7] = q1.w;
            acc = __builtin_amdgcn_wmma_f32_16x16x32_bf16(
                false, a[kt].v, false, b.v, (short)0, acc, false, false);
        }
        int ncol   = nt * 16 + m;
        float bias = hidden_b[conv * GIN_H + ncol];
        #pragma unroll
        for (int r = 0; r < 8; ++r) {
            float val = fmaxf(acc[r] + bias, 0.f);
            hsh[wave][r + half * 8][ncol] = f2bf(val);
        }
    }

    // ---- GEMM2: 4 N-tiles x 8 K-steps, A streamed from LDS as b128 ----
    #pragma unroll
    for (int nt = 0; nt < 4; ++nt) {
        v8f acc = {0.f, 0.f, 0.f, 0.f, 0.f, 0.f, 0.f, 0.f};
        #pragma unroll
        for (int kt = 0; kt < 8; ++kt) {
            BF16x16 a2, b;
            uint4 h0 = *(const uint4*)&hsh[wave][m][kt * 32 + half * 8];
            uint4 h1 = *(const uint4*)&hsh[wave][m][kt * 32 + 16 + half * 8];
            a2.u[0] = h0.x; a2.u[1] = h0.y; a2.u[2] = h0.z; a2.u[3] = h0.w;
            a2.u[4] = h1.x; a2.u[5] = h1.y; a2.u[6] = h1.z; a2.u[7] = h1.w;
            const uint4* p = (const uint4*)(w2p + ((((conv * 4 + nt) * 8 + kt) << 8) + lane * 8));
            uint4 q0 = p[0], q1 = p[1];
            b.u[0] = q0.x; b.u[1] = q0.y; b.u[2] = q0.z; b.u[3] = q0.w;
            b.u[4] = q1.x; b.u[5] = q1.y; b.u[6] = q1.z; b.u[7] = q1.w;
            acc = __builtin_amdgcn_wmma_f32_16x16x32_bf16(
                false, a2.v, false, b.v, (short)0, acc, false, false);
        }
        int ncol   = nt * 16 + m;
        float bias = out_b[conv * CHANNELS + ncol];
        float s = 0.f, s2 = 0.f;
        #pragma unroll
        for (int r = 0; r < 8; ++r) {
            float val = acc[r] + bias;
            cbuf[(size_t)(row0 + r + half * 8) * CHANNELS + ncol] = val;
            s  += val;
            s2 += val * val;
        }
        atomicAdd(&stats[ncol], s);
        atomicAdd(&stats[CHANNELS + ncol], s2);
    }
}

__global__ void stats_finalize_kernel(float* __restrict__ stats) {
    int c = threadIdx.x;            // 64 threads
    float inv_n = 1.0f / (float)N_NODES;
    float mu  = stats[c] * inv_n;
    float var = stats[64 + c] * inv_n - mu * mu;
    stats[128 + c] = mu;
    stats[192 + c] = rsqrtf(var + 1e-5f);
}

// x_{i+1} = relu(x_i + (c - mu) * rsqrt(var+eps))  -> next column block of xcat
__global__ void norm_residual_kernel(const float* __restrict__ cbuf,
                                     const float* __restrict__ stats,
                                     float* __restrict__ xcat, int conv) {
    int idx = blockIdx.x * blockDim.x + threadIdx.x;
    if (idx >= N_NODES * CHANNELS) return;
    int n = idx >> 6, c = idx & 63;
    float mu  = stats[128 + c];
    float inv = stats[192 + c];
    float xv  = xcat[(size_t)n * XCAT_W + conv * CHANNELS + c];
    float val = fmaxf(xv + (cbuf[idx] - mu) * inv, 0.f);
    xcat[(size_t)n * XCAT_W + (conv + 1) * CHANNELS + c] = val;
}

// segment_max via monotonic uint encoding + atomicMax
__global__ void pool_max_kernel(const int* __restrict__ batch,
                                const float* __restrict__ xcat,
                                unsigned int* __restrict__ pooledU) {
    int idx = blockIdx.x * blockDim.x + threadIdx.x;
    if (idx >= N_NODES * XCAT_W) return;
    int n = idx / XCAT_W;
    int c = idx - n * XCAT_W;
    int g = batch[n];
    unsigned b = __float_as_uint(xcat[idx]);
    unsigned u = (b & 0x80000000u) ? ~b : (b | 0x80000000u);
    atomicMax(&pooledU[(size_t)g * XCAT_W + c], u);
}

__global__ void pool_decode_kernel(const unsigned int* __restrict__ pooledU,
                                   float* __restrict__ pooledF) {
    int idx = blockIdx.x * blockDim.x + threadIdx.x;
    if (idx >= N_GRAPHS * XCAT_W) return;
    unsigned u = pooledU[idx];
    unsigned b = (u & 0x80000000u) ? (u & 0x7FFFFFFFu) : ~u;
    float f = __uint_as_float(b);
    if (((b >> 23) & 0xFFu) == 0xFFu) f = 0.f;   // empty segment / non-finite -> 0
    pooledF[idx] = f;
}

// h2 = relu(pooled @ final_w + final_b)   [256x576]@[576x1024]
__global__ void final_dense_kernel(const float* __restrict__ pooledF,
                                   const float* __restrict__ final_w,
                                   const float* __restrict__ final_b,
                                   float* __restrict__ h2) {
    int idx = blockIdx.x * blockDim.x + threadIdx.x;
    if (idx >= N_GRAPHS * HIDDEN) return;
    int g = idx >> 10, o = idx & 1023;
    float s = final_b[o];
    const float* pr = pooledF + (size_t)g * XCAT_W;
    for (int k = 0; k < XCAT_W; ++k)
        s += pr[k] * final_w[(size_t)k * HIDDEN + o];
    h2[idx] = fmaxf(s, 0.f);
}

__global__ void cls_kernel(const float* __restrict__ h2,
                           const float* __restrict__ cls_w,
                           const float* __restrict__ cls_b,
                           float* __restrict__ out) {
    int g = threadIdx.x;     // 256 threads, 1 block
    float s = cls_b[0];
    const float* hr = h2 + (size_t)g * HIDDEN;
    for (int k = 0; k < HIDDEN; ++k) s += hr[k] * cls_w[k];
    out[g] = s;
}

extern "C" void kernel_launch(void* const* d_in, const int* in_sizes, int n_in,
                              void* d_out, int out_size, void* d_ws, size_t ws_size,
                              hipStream_t stream) {
    const int*   node_types = (const int*)  d_in[0];
    const int*   edge_index = (const int*)  d_in[1];
    const int*   batch      = (const int*)  d_in[2];
    const float* emb        = (const float*)d_in[3];
    const float* hidden_w   = (const float*)d_in[4];
    const float* hidden_b   = (const float*)d_in[5];
    const float* out_w      = (const float*)d_in[6];
    const float* out_b      = (const float*)d_in[7];
    const float* final_w    = (const float*)d_in[8];
    const float* final_b    = (const float*)d_in[9];
    const float* cls_w      = (const float*)d_in[10];
    const float* cls_b      = (const float*)d_in[11];
    float* out = (float*)d_out;

    char* ws = (char*)d_ws;
    size_t off = 0;
    auto take = [&](size_t bytes) -> void* {
        void* p = ws + off;
        off += (bytes + 255) & ~(size_t)255;
        return p;
    };
    const size_t XCAT_B = (size_t)N_NODES * XCAT_W * 4;      // 115.2 MB
    const size_t AGG_B  = (size_t)N_NODES * 128 * 4;         // 25.6 MB
    float*        xcat    = (float*)take(XCAT_B);
    float*        agg     = (float*)take(AGG_B + 1024);      // stats ride behind agg
    float*        stats   = agg + (AGG_B / 4);               // sum|sumsq|mu|inv (4x64)
    float*        cbuf    = (float*)take((size_t)N_NODES * CHANNELS * 4);
    unsigned int* w1p     = (unsigned int*)take((size_t)CONVS * 16 * 4 * 256 * 4);
    unsigned int* w2p     = (unsigned int*)take((size_t)CONVS * 4 * 8 * 256 * 4);
    unsigned int* pooledU = (unsigned int*)take((size_t)N_GRAPHS * XCAT_W * 4);
    float*        pooledF = (float*)take((size_t)N_GRAPHS * XCAT_W * 4);
    float*        h2      = (float*)take((size_t)N_GRAPHS * HIDDEN * 4);

    pack_weights_kernel<<<(CONVS * 16 * 4 * 256 + CONVS * 4 * 8 * 256 + 255) / 256, 256, 0, stream>>>(
        hidden_w, out_w, w1p, w2p);
    embed_kernel<<<(N_NODES * CHANNELS + 255) / 256, 256, 0, stream>>>(node_types, emb, xcat);

    for (int conv = 0; conv < CONVS; ++conv) {
        hipMemsetAsync(agg, 0, AGG_B + 1024, stream);
        scatter_kernel<<<(N_EDGES * 16 + 255) / 256, 256, 0, stream>>>(edge_index, xcat, agg, conv);
        conv_gemm_kernel<<<(M_TILES + 3) / 4, 128, 0, stream>>>(
            agg, w1p, w2p, hidden_b, out_b, cbuf, stats, conv);
        stats_finalize_kernel<<<1, 64, 0, stream>>>(stats);
        norm_residual_kernel<<<(N_NODES * CHANNELS + 255) / 256, 256, 0, stream>>>(
            cbuf, stats, xcat, conv);
    }

    hipMemsetAsync(pooledU, 0, (size_t)N_GRAPHS * XCAT_W * 4, stream);
    pool_max_kernel<<<(N_NODES * XCAT_W + 255) / 256, 256, 0, stream>>>(batch, xcat, pooledU);
    pool_decode_kernel<<<(N_GRAPHS * XCAT_W + 255) / 256, 256, 0, stream>>>(pooledU, pooledF);
    final_dense_kernel<<<(N_GRAPHS * HIDDEN + 255) / 256, 256, 0, stream>>>(
        pooledF, final_w, final_b, h2);
    cls_kernel<<<1, 256, 0, stream>>>(h2, cls_w, cls_b, out);
}
